// GNNPolicyValueNet_20796231647363
// MI455X (gfx1250) — compile-verified, matrix operand order checked by reading, not run on hardware
//
#include <hip/hip_runtime.h>
#include <hip/hip_bf16.h>

typedef _Float16 v16h __attribute__((ext_vector_type(16)));
typedef _Float16 v8h  __attribute__((ext_vector_type(8)));
typedef float    v8f  __attribute__((ext_vector_type(8)));

#define H 256
#define GNUM 256

// ---------------- utility kernels ----------------

__global__ __launch_bounds__(256) void k_zero_f32(float* __restrict__ p, long n) {
    long t = (long)blockIdx.x * blockDim.x + threadIdx.x;
    if (t < n) p[t] = 0.0f;
}

__global__ __launch_bounds__(256) void k_f32_to_f16(const float* __restrict__ s,
                                                    _Float16* __restrict__ d, long n) {
    long t = (long)blockIdx.x * blockDim.x + threadIdx.x;
    if (t < n) d[t] = (_Float16)s[t];
}

// Pack a 256x256 f32 row-major weight into WMMA B-fragment order:
// Bp[((ct*8 + ks)*32 + lane)*16 + i], so the GEMM reads one v16h per lane per K-step.
__global__ __launch_bounds__(256) void k_pack_B(const float* __restrict__ W,
                                                _Float16* __restrict__ Bp) {
    int t = blockIdx.x * blockDim.x + threadIdx.x;   // 0 .. 65535
    int i    = t & 15;
    int lane = (t >> 4) & 31;
    int ks   = (t >> 9) & 7;
    int ct   = t >> 12;
    int hi   = lane >> 4;
    int n    = ct * 16 + (lane & 15);
    int K    = ks * 32 + (i < 8 ? hi * 8 + i : 16 + hi * 8 + (i - 8));
    Bp[t] = (_Float16)W[K * H + n];
}

// ---------------- WMMA GEMM: C[M x 256] = A[M x 256] * W + bias (opt relu) -------------
// A: f16 row-major. Bp: packed B fragments.
// One wave -> 16x64 output (4 col tiles), sharing one A fragment per K-step.
// Block = 8 waves = 2 row blocks x 4 col-quarters. Grid.x = M/32.
__global__ __launch_bounds__(256)
void k_gemm_wmma(const _Float16* __restrict__ A,
                 const _Float16* __restrict__ Bp,
                 const float* __restrict__ bias,   // [256] or nullptr
                 float* __restrict__ C,
                 int M, int relu)
{
    const int wave = threadIdx.x >> 5;          // 0..7
    const int lane = threadIdx.x & 31;
    const int r    = lane & 15;
    const int hi   = lane >> 4;
    const int row0 = (blockIdx.x * 2 + (wave >> 2)) * 16;
    const int ct0  = (wave & 3) * 4;            // 4 consecutive col tiles
    if (row0 >= M) return;

    v8f acc[4] = {};
    const _Float16* arow = A + (size_t)(row0 + r) * H;

    #pragma unroll
    for (int ks = 0; ks < 8; ++ks) {
        const int k0 = ks * 32;
        v8h a_lo = *(const v8h*)(arow + k0 + hi * 8);
        v8h a_hi = *(const v8h*)(arow + k0 + 16 + hi * 8);
        v16h a = __builtin_shufflevector(a_lo, a_hi,
                 0,1,2,3,4,5,6,7, 8,9,10,11,12,13,14,15);
        #pragma unroll
        for (int c = 0; c < 4; ++c) {
            const int ct = ct0 + c;
            v16h b = *(const v16h*)(Bp + (((size_t)(ct * 8 + ks) * 32 + lane) * 16));
            acc[c] = __builtin_amdgcn_wmma_f32_16x16x32_f16(
                         false, a, false, b, (short)0, acc[c], false, false);
        }
    }

    #pragma unroll
    for (int c = 0; c < 4; ++c) {
        const int col0 = (ct0 + c) * 16;
        const float bv = bias ? bias[col0 + r] : 0.0f;
        #pragma unroll
        for (int j = 0; j < 8; ++j) {
            int row = row0 + hi * 8 + j;
            float v = acc[c][j] + bv;
            if (relu) v = fmaxf(v, 0.0f);
            C[(size_t)row * H + col0 + r] = v;
        }
    }
}

// ---------------- GCN normalization ----------------

__global__ __launch_bounds__(256) void k_deg_init(float* __restrict__ deg, int n) {
    int t = blockIdx.x * blockDim.x + threadIdx.x;
    if (t < n) deg[t] = 1.0f;   // self-loop
}

__global__ __launch_bounds__(256) void k_deg_edges(const int* __restrict__ col,
                                                   float* __restrict__ deg, long E) {
    long t = (long)blockIdx.x * blockDim.x + threadIdx.x;
    if (t < E) atomicAdd(&deg[col[t]], 1.0f);
}

__global__ __launch_bounds__(256) void k_dinv(float* __restrict__ deg, int n) {
    int t = blockIdx.x * blockDim.x + threadIdx.x;
    if (t < n) { float d = deg[t]; deg[t] = (d > 0.0f) ? rsqrtf(d) : 0.0f; }
}

// one block per edge: coalesced gather of hB[row], coalesced atomic scatter to hA[col]
__global__ __launch_bounds__(256)
void k_scatter(const int* __restrict__ rowI, const int* __restrict__ colI,
               const float* __restrict__ dinv,
               const float* __restrict__ hB, float* __restrict__ hA)
{
    const int e = blockIdx.x;
    const int f = threadIdx.x;
    const int rs = rowI[e];
    const int cs = colI[e];
    const float norm = dinv[rs] * dinv[cs];
    atomicAdd(&hA[(size_t)cs * H + f], hB[(size_t)rs * H + f] * norm);
}

// h = relu(scatterAccum + hB*dinv^2 + bias); also emit f16 copy for the next GEMM
__global__ __launch_bounds__(256)
void k_finalize(float* __restrict__ hA, const float* __restrict__ hB,
                const float* __restrict__ dinv, const float* __restrict__ bias,
                _Float16* __restrict__ hF, long n)
{
    long t = (long)blockIdx.x * blockDim.x + threadIdx.x;
    if (t >= n) return;
    const int i = (int)(t >> 8);
    const int f = (int)(t & 255);
    const float d = dinv[i];
    float v = hA[t] + hB[t] * d * d + bias[f];
    v = fmaxf(v, 0.0f);
    hA[t] = v;
    hF[t] = (_Float16)v;
}

// ---------------- pooling + heads ----------------

__global__ __launch_bounds__(256)
void k_pool(const float* __restrict__ hA, const int* __restrict__ batch,
            float* __restrict__ pooled, float* __restrict__ cnts)
{
    const int i = blockIdx.x;
    const int f = threadIdx.x;
    const int g = batch[i];
    atomicAdd(&pooled[(size_t)g * H + f], hA[(size_t)i * H + f]);
    if (f == 0) atomicAdd(&cnts[g], 1.0f);
}

__global__ __launch_bounds__(256)
void k_pool_div(float* __restrict__ pooled, const float* __restrict__ cnts) {
    const int g = blockIdx.x;
    const int f = threadIdx.x;
    pooled[(size_t)g * H + f] /= fmaxf(cnts[g], 1.0f);
}

// policy logits: out[i] = dot(t[i,:], Wp2) + bp2 ; one wave per node
__global__ __launch_bounds__(256)
void k_policy(const float* __restrict__ t, const float* __restrict__ Wp2,
              const float* __restrict__ bp2, float* __restrict__ out, int N)
{
    const int wave = threadIdx.x >> 5;
    const int lane = threadIdx.x & 31;
    const int i = blockIdx.x * 8 + wave;
    if (i >= N) return;
    const float* r = t + (size_t)i * H;
    float s = 0.0f;
    #pragma unroll
    for (int k = 0; k < 8; ++k) s += r[lane + 32 * k] * Wp2[lane + 32 * k];
    #pragma unroll
    for (int m = 16; m >= 1; m >>= 1) s += __shfl_xor(s, m, 32);
    if (lane == 0) out[i] = s + bp2[0];
}

// value hidden: t2[g,j] = relu(sum_k pooled[g,k]*Wv1[k,j] + bv1[j])
__global__ __launch_bounds__(256)
void k_value_h(const float* __restrict__ pooled, const float* __restrict__ Wv1,
               const float* __restrict__ bv1, float* __restrict__ t2)
{
    const int g = blockIdx.x;
    const int j = threadIdx.x;
    float s = 0.0f;
    for (int k = 0; k < H; ++k) s += pooled[(size_t)g * H + k] * Wv1[(size_t)k * H + j];
    t2[(size_t)g * H + j] = fmaxf(s + bv1[j], 0.0f);
}

__global__ __launch_bounds__(256)
void k_value_out(const float* __restrict__ t2, const float* __restrict__ Wv2,
                 const float* __restrict__ bv2, float* __restrict__ out)
{
    const int g = threadIdx.x;   // 256 graphs, single block
    float s = 0.0f;
    for (int k = 0; k < H; ++k) s += t2[(size_t)g * H + k] * Wv2[k];
    out[g] = tanhf(s + bv2[0]);
}

// ---------------- launcher ----------------

static inline long cdivl(long a, long b) { return (a + b - 1) / b; }

extern "C" void kernel_launch(void* const* d_in, const int* in_sizes, int n_in,
                              void* d_out, int out_size, void* d_ws, size_t ws_size,
                              hipStream_t stream) {
    const float* x        = (const float*)d_in[0];
    const int*   ei       = (const int*)d_in[1];
    const int    E        = in_sizes[2];            // edge_attr count
    const int*   batch    = (const int*)d_in[3];
    const int    N        = in_sizes[3];
    const float* W_enc    = (const float*)d_in[5];
    const float* b_enc    = (const float*)d_in[6];
    const float* W_convs  = (const float*)d_in[8];
    const float* b_convs  = (const float*)d_in[9];
    const float* Wp1      = (const float*)d_in[10];
    const float* bp1      = (const float*)d_in[11];
    const float* Wp2      = (const float*)d_in[12];
    const float* bp2      = (const float*)d_in[13];
    const float* Wv1      = (const float*)d_in[14];
    const float* bv1      = (const float*)d_in[15];
    const float* Wv2      = (const float*)d_in[16];
    const float* bv2      = (const float*)d_in[17];
    float* out = (float*)d_out;

    const long NH = (long)N * H;

    // workspace carve (256B aligned)
    char* p = (char*)d_ws;
    auto carve = [&](size_t bytes) {
        char* q = p;
        p += (bytes + 255) & ~(size_t)255;
        return (void*)q;
    };
    float*    hA     = (float*)carve(NH * sizeof(float));
    float*    hB     = (float*)carve(NH * sizeof(float));
    _Float16* hF     = (_Float16*)carve(NH * sizeof(_Float16));
    _Float16* wF     = (_Float16*)carve((size_t)H * H * sizeof(_Float16));
    float*    dinv   = (float*)carve((size_t)N * sizeof(float));
    float*    pooled = (float*)carve((size_t)GNUM * H * sizeof(float));
    float*    cnts   = (float*)carve((size_t)GNUM * sizeof(float));
    float*    t2     = (float*)carve((size_t)GNUM * H * sizeof(float));

    const int* rowI = ei;
    const int* colI = ei + E;

    const dim3 blk(256);
    const dim3 gemmGrid((unsigned)cdivl(N, 32));  // N = 100000 -> 3125 blocks
    const int  packBlocks = (H * H) / 256;

    // ---- encoder: h = x @ W_enc + b_enc ----
    k_pack_B<<<packBlocks, blk, 0, stream>>>(W_enc, wF);
    k_f32_to_f16<<<(int)cdivl(NH, 256), blk, 0, stream>>>(x, hF, NH);
    k_gemm_wmma<<<gemmGrid, blk, 0, stream>>>(hF, wF, b_enc, hA, N, 0);
    k_f32_to_f16<<<(int)cdivl(NH, 256), blk, 0, stream>>>(hA, hF, NH);

    // ---- GCN normalization: dinv = rsqrt(deg_in + selfloop) ----
    k_deg_init<<<(int)cdivl(N, 256), blk, 0, stream>>>(dinv, N);
    k_deg_edges<<<(int)cdivl(E, 256), blk, 0, stream>>>(colI, dinv, E);
    k_dinv<<<(int)cdivl(N, 256), blk, 0, stream>>>(dinv, N);

    // ---- 3 GCN conv layers ----
    for (int l = 0; l < 3; ++l) {
        k_pack_B<<<packBlocks, blk, 0, stream>>>(W_convs + (size_t)l * H * H, wF);
        k_gemm_wmma<<<gemmGrid, blk, 0, stream>>>(hF, wF, nullptr, hB, N, 0);
        k_zero_f32<<<(int)cdivl(NH, 256), blk, 0, stream>>>(hA, NH);
        k_scatter<<<E, blk, 0, stream>>>(rowI, colI, dinv, hB, hA);
        k_finalize<<<(int)cdivl(NH, 256), blk, 0, stream>>>(
            hA, hB, dinv, b_convs + (size_t)l * H, hF, NH);
    }

    // ---- global mean pool ----
    k_zero_f32<<<(int)cdivl((long)GNUM * H + GNUM, 256), blk, 0, stream>>>(pooled,
                                                                (long)GNUM * H + GNUM);
    k_pool<<<N, blk, 0, stream>>>(hA, batch, pooled, cnts);
    k_pool_div<<<GNUM, blk, 0, stream>>>(pooled, cnts);

    // ---- policy head ----
    k_pack_B<<<packBlocks, blk, 0, stream>>>(Wp1, wF);
    k_gemm_wmma<<<gemmGrid, blk, 0, stream>>>(hF, wF, bp1, hB, N, 1);
    k_policy<<<(int)cdivl(N, 8), blk, 0, stream>>>(hB, Wp2, bp2, out, N);

    // ---- value head ----
    k_value_h<<<GNUM, blk, 0, stream>>>(pooled, Wv1, bv1, t2);
    k_value_out<<<1, blk, 0, stream>>>(t2, Wv2, bv2, out + N);

    (void)n_in; (void)out_size; (void)ws_size;
}
